// MultilayerGRU_2559800508689
// MI455X (gfx1250) — matrix operand order repaired and checked by reading, not compiled, not run on hardware
//
#include <hip/hip_runtime.h>
#include <cstdint>
#include <cstddef>

// ---------------------------------------------------------------------------
// MultilayerGRU on gfx1250 (MI455X).
// bf16 WMMA (v_wmma_f32_16x16x32_bf16), fp32 accumulate.
// Whole 1024-step scan in ONE persistent kernel (64 WGs x 128 thr, 256 waves,
// one 16x16 tile per wave) with device-side grid barriers.
// No __restrict__ inside the scan kernel: the barrier's asm memory clobber is
// then a true fence (no cross-barrier load hoisting -> correct vs other WGs'
// stores, and no giant weight-register caching). GEMM helpers are chunked,
// double-buffered pipelines: one amortized memory latency per 128-K chunk.
// ---------------------------------------------------------------------------

static constexpr int B_ = 64;     // batch
static constexpr int S_ = 1024;   // sequence
static constexpr int I_ = 256;    // input dim
static constexpr int H_ = 1024;   // hidden dim
static constexpr int O_ = 256;    // output dim
static constexpr int NWG_ = 64;   // persistent workgroups (64 x 4 waves = 256 tiles)

typedef __bf16 v8bf  __attribute__((ext_vector_type(8)));
typedef __bf16 v16bf __attribute__((ext_vector_type(16)));
typedef float  v8f   __attribute__((ext_vector_type(8)));

// ---------------- WMMA fragment helpers (ISA 7.12.2 layouts) ----------------

// A: 16x32 bf16 (MxK). lane m in [0,15] holds row m, K = k0..k0+7 / k0+16..23;
// lane m+16 holds row m, K = k0+8..15 / k0+24..31.
__device__ __forceinline__ v16bf load_a_frag(const __bf16* A, size_t lda,
                                             int m0, int k0) {
  const int lane = threadIdx.x & 31;
  const __bf16* p = A + (size_t)(m0 + (lane & 15)) * lda
                      + (size_t)(k0 + ((lane >> 4) << 3));
  v8bf lo = *reinterpret_cast<const v8bf*>(p);
  v8bf hi = *reinterpret_cast<const v8bf*>(p + 16);
  v16bf r;
#pragma unroll
  for (int i = 0; i < 8; ++i) { r[i] = lo[i]; r[i + 8] = hi[i]; }
  return r;
}

// B: 32x16 bf16 (KxN), B[k][n] = W[n][k] with W row-major (N x K).
// lane n in [0,15]: column n, K = k0..k0+7 / k0+16..23 (in W row n);
// lane n+16: column n, K = k0+8..15 / k0+24..31.
__device__ __forceinline__ v16bf load_b_frag(const __bf16* W, size_t ldw,
                                             int n0, int k0) {
  const int lane = threadIdx.x & 31;
  const __bf16* p = W + (size_t)(n0 + (lane & 15)) * ldw
                      + (size_t)(k0 + ((lane >> 4) << 3));
  v8bf lo = *reinterpret_cast<const v8bf*>(p);
  v8bf hi = *reinterpret_cast<const v8bf*>(p + 16);
  v16bf r;
#pragma unroll
  for (int i = 0; i < 8; ++i) { r[i] = lo[i]; r[i + 8] = hi[i]; }
  return r;
}

__device__ __forceinline__ v8f wmma_bf16(v16bf a, v16bf b, v8f c) {
  return __builtin_amdgcn_wmma_f32_16x16x32_bf16(
      false, a, false, b, (short)0, c, false, false);
}

__device__ __forceinline__ float sigmoidf_(float x) {
  return 1.0f / (1.0f + __expf(-x));
}

// Chunked, double-buffered GEMM: CH k-steps (32 K each) per chunk.
// Loads for chunk c+1 are issued (one big clause) before chunk c's WMMAs;
// K is a template parameter so everything fully unrolls (SSA renaming ->
// real rotating buffers, partial s_wait_loadcnt instead of full stalls).

template <int K>
__device__ __forceinline__ void gemm_acc(v8f& acc,
                                         const __bf16* A, size_t lda,
                                         const __bf16* W, size_t ldw,
                                         int m0, int n0) {
  constexpr int CH = 4;
  constexpr int NC = K / (32 * CH);
  static_assert(K % (32 * CH) == 0, "K must be a multiple of 128");
  v16bf a[CH], b[CH];
#pragma unroll
  for (int i = 0; i < CH; ++i) {
    a[i] = load_a_frag(A, lda, m0, i * 32);
    b[i] = load_b_frag(W, ldw, n0, i * 32);
  }
#pragma unroll
  for (int c = 1; c < NC; ++c) {
    v16bf an[CH], bn[CH];
#pragma unroll
    for (int i = 0; i < CH; ++i) {
      const int k0 = (c * CH + i) * 32;
      an[i] = load_a_frag(A, lda, m0, k0);
      bn[i] = load_b_frag(W, ldw, n0, k0);
    }
#pragma unroll
    for (int i = 0; i < CH; ++i) acc = wmma_bf16(a[i], b[i], acc);
#pragma unroll
    for (int i = 0; i < CH; ++i) { a[i] = an[i]; b[i] = bn[i]; }
  }
#pragma unroll
  for (int i = 0; i < CH; ++i) acc = wmma_bf16(a[i], b[i], acc);
}

// two accumulators sharing the A fragment (z and r gates)
template <int K>
__device__ __forceinline__ void gemm_acc2(v8f& az, v8f& ar,
                                          const __bf16* A, size_t lda,
                                          const __bf16* Wz, const __bf16* Wr,
                                          size_t ldw, int m0, int n0) {
  constexpr int CH = 4;
  constexpr int NC = K / (32 * CH);
  static_assert(K % (32 * CH) == 0, "K must be a multiple of 128");
  v16bf a[CH], bz[CH], br[CH];
#pragma unroll
  for (int i = 0; i < CH; ++i) {
    a[i]  = load_a_frag(A, lda, m0, i * 32);
    bz[i] = load_b_frag(Wz, ldw, n0, i * 32);
    br[i] = load_b_frag(Wr, ldw, n0, i * 32);
  }
#pragma unroll
  for (int c = 1; c < NC; ++c) {
    v16bf an[CH], bzn[CH], brn[CH];
#pragma unroll
    for (int i = 0; i < CH; ++i) {
      const int k0 = (c * CH + i) * 32;
      an[i]  = load_a_frag(A, lda, m0, k0);
      bzn[i] = load_b_frag(Wz, ldw, n0, k0);
      brn[i] = load_b_frag(Wr, ldw, n0, k0);
    }
#pragma unroll
    for (int i = 0; i < CH; ++i) {
      az = wmma_bf16(a[i], bz[i], az);
      ar = wmma_bf16(a[i], br[i], ar);
    }
#pragma unroll
    for (int i = 0; i < CH; ++i) { a[i] = an[i]; bz[i] = bzn[i]; br[i] = brn[i]; }
  }
#pragma unroll
  for (int i = 0; i < CH; ++i) {
    az = wmma_bf16(a[i], bz[i], az);
    ar = wmma_bf16(a[i], br[i], ar);
  }
}

// Monotonic-counter grid barrier (device scope), s_sleep spin.
// With no noalias pointers in the scan kernel, the asm memory clobbers are
// hard fences: no load crosses a barrier, nothing hoists out of the t loop.
__device__ __forceinline__ void grid_barrier(unsigned* cnt, int& episode) {
  episode += 1;
  asm volatile("" ::: "memory");
  __threadfence();
  __syncthreads();
  if (threadIdx.x == 0) {
    __hip_atomic_fetch_add(cnt, 1u, __ATOMIC_ACQ_REL, __HIP_MEMORY_SCOPE_AGENT);
    const unsigned target = (unsigned)episode * (unsigned)NWG_;
    while (__hip_atomic_load(cnt, __ATOMIC_ACQUIRE, __HIP_MEMORY_SCOPE_AGENT)
           < target) {
      __builtin_amdgcn_s_sleep(1);
    }
  }
  __syncthreads();
  __threadfence();
  asm volatile("" ::: "memory");
}

// ---------------------------- utility kernels -------------------------------

__global__ void __launch_bounds__(256) k_f32_to_bf16(const float* __restrict__ s,
                                                     __bf16* __restrict__ d,
                                                     size_t n) {
  size_t i = (size_t)blockIdx.x * 256 + threadIdx.x;
  if (i < n) d[i] = (__bf16)s[i];
}

__global__ void __launch_bounds__(64) k_zero_u32(unsigned* __restrict__ p) {
  if (threadIdx.x == 0) *p = 0u;
}

// hidden_state (B, 2, H) -> per-layer fp32 + bf16 state buffers
__global__ void __launch_bounds__(256) k_init_h(const float* __restrict__ hs,
                                                float* __restrict__ h0f,
                                                float* __restrict__ h1f,
                                                __bf16* __restrict__ h0b,
                                                __bf16* __restrict__ h1b) {
  int i = blockIdx.x * 256 + threadIdx.x;
  if (i >= B_ * H_) return;
  int b = i >> 10, h = i & (H_ - 1);
  float a0 = hs[(size_t)b * 2 * H_ + h];
  float a1 = hs[(size_t)b * 2 * H_ + H_ + h];
  h0f[i] = a0; h1f[i] = a1;
  h0b[i] = (__bf16)a0; h1b[i] = (__bf16)a1;
}

// hidden_out flat == concat([h0, h1]) flat (reshape keeps flat order)
__global__ void __launch_bounds__(256) k_copy_hidden(const float* __restrict__ h0,
                                                     const float* __restrict__ h1,
                                                     float* __restrict__ out) {
  int i = blockIdx.x * 256 + threadIdx.x;
  if (i >= 2 * B_ * H_) return;
  out[i] = (i < B_ * H_) ? h0[i] : h1[i - B_ * H_];
}

// ----------------------- persistent GRU scan kernel -------------------------
// Tile map: wave id = blockIdx.x*4 + wave; m0 = (id&3)*16 (batch),
// n0 = (id>>2)*16 (hidden col). 4 waves of a WG share n0 -> weight lines hit
// in the WGP$; L2 sees each weight matrix ~once per step (L2-resident).
// NOTE: no __restrict__ on these params (correctness: other WGs write the
// state buffers between barriers; noalias would license hoisting past them).

__global__ void __launch_bounds__(128) k_gru_scan(
    const __bf16* Xbf,
    const __bf16* wxz0, const __bf16* whz0, const float* __restrict__ bz0,
    const __bf16* wxr0, const __bf16* whr0, const float* __restrict__ br0,
    const __bf16* wxg0, const __bf16* whg0, const float* __restrict__ bg0,
    const __bf16* wxz1, const __bf16* whz1, const float* __restrict__ bz1,
    const __bf16* wxr1, const __bf16* whr1, const float* __restrict__ br1,
    const __bf16* wxg1, const __bf16* whg1, const float* __restrict__ bg1,
    float* h0f, float* h1f,
    __bf16* h0b, __bf16* h1b,
    float* zf0, float* zf1,
    __bf16* rh0, __bf16* rh1,
    __bf16* hist, unsigned* bar) {
  const int wave = threadIdx.x >> 5;
  const int id = blockIdx.x * 4 + wave;          // 0..255
  const int m0 = (id & 3) * 16;                  // batch tile
  const int n0 = (id >> 2) * 16;                 // hidden tile
  const int lane = threadIdx.x & 31;
  const int n = n0 + (lane & 15);
  const int mlo = m0 + ((lane >> 4) << 3);       // C/D layout row base
  int episode = 0;

  const float vbz0 = bz0[n], vbr0 = br0[n], vbg0 = bg0[n];
  const float vbz1 = bz1[n], vbr1 = br1[n], vbg1 = bg1[n];
  const size_t ldx = (size_t)S_ * I_;

#pragma clang loop unroll(disable)
  for (int t = 0; t < S_; ++t) {
    const __bf16* Xt = Xbf + (size_t)t * I_;

    // ---- phase A, layer 0: z0, r0 ----
    {
      v8f az, ar;
#pragma unroll
      for (int j = 0; j < 8; ++j) { az[j] = vbz0; ar[j] = vbr0; }
      gemm_acc2<I_>(az, ar, Xt, ldx, wxz0, wxr0, (size_t)I_, m0, n0);
      gemm_acc2<H_>(az, ar, h0b, (size_t)H_, whz0, whr0, (size_t)H_, m0, n0);
#pragma unroll
      for (int j = 0; j < 8; ++j) {
        const size_t idx = (size_t)(mlo + j) * H_ + n;
        zf0[idx] = sigmoidf_(az[j]);
        rh0[idx] = (__bf16)(sigmoidf_(ar[j]) * h0f[idx]);
      }
    }
    grid_barrier(bar, episode);

    // ---- phase B, layer 0: g0, h0 update ----
    {
      v8f ag;
#pragma unroll
      for (int j = 0; j < 8; ++j) ag[j] = vbg0;
      gemm_acc<I_>(ag, Xt, ldx, wxg0, (size_t)I_, m0, n0);
      gemm_acc<H_>(ag, rh0, (size_t)H_, whg0, (size_t)H_, m0, n0);
#pragma unroll
      for (int j = 0; j < 8; ++j) {
        const size_t idx = (size_t)(mlo + j) * H_ + n;
        const float g = tanhf(ag[j]);
        const float z = zf0[idx];
        const float hn = z * h0f[idx] + (1.0f - z) * g;
        h0f[idx] = hn;
        h0b[idx] = (__bf16)hn;
      }
    }
    grid_barrier(bar, episode);

    // ---- phase A, layer 1 (x = new h0): z1, r1 ----
    {
      v8f az, ar;
#pragma unroll
      for (int j = 0; j < 8; ++j) { az[j] = vbz1; ar[j] = vbr1; }
      gemm_acc2<H_>(az, ar, h0b, (size_t)H_, wxz1, wxr1, (size_t)H_, m0, n0);
      gemm_acc2<H_>(az, ar, h1b, (size_t)H_, whz1, whr1, (size_t)H_, m0, n0);
#pragma unroll
      for (int j = 0; j < 8; ++j) {
        const size_t idx = (size_t)(mlo + j) * H_ + n;
        zf1[idx] = sigmoidf_(az[j]);
        rh1[idx] = (__bf16)(sigmoidf_(ar[j]) * h1f[idx]);
      }
    }
    grid_barrier(bar, episode);

    // ---- phase B, layer 1: g1, h1 update + history ----
    {
      v8f ag;
#pragma unroll
      for (int j = 0; j < 8; ++j) ag[j] = vbg1;
      gemm_acc<H_>(ag, h0b, (size_t)H_, wxg1, (size_t)H_, m0, n0);
      gemm_acc<H_>(ag, rh1, (size_t)H_, whg1, (size_t)H_, m0, n0);
#pragma unroll
      for (int j = 0; j < 8; ++j) {
        const int m = mlo + j;
        const size_t idx = (size_t)m * H_ + n;
        const float g = tanhf(ag[j]);
        const float z = zf1[idx];
        const float hn = z * h1f[idx] + (1.0f - z) * g;
        h1f[idx] = hn;
        const __bf16 hb = (__bf16)hn;
        h1b[idx] = hb;
        hist[((size_t)t * B_ + m) * H_ + n] = hb;
      }
    }
    grid_barrier(bar, episode);
  }
}

// Batched output projection: y[b,s,:] = h1_hist[s,b,:] @ Wy^T + by.
// M = S*B = 65536 rows, N = O = 256, K = H. One 16x16 tile per wave.
__global__ void __launch_bounds__(128) k_out_proj(
    const __bf16* hist, const __bf16* Wy,
    const float* __restrict__ by, float* out) {
  const int wave = threadIdx.x >> 5;
  const int id = blockIdx.x * 4 + wave;          // 0 .. 65535
  const int n0 = (id & 15) * 16;
  const int m0 = (id >> 4) * 16;
  const int lane = threadIdx.x & 31;
  const int n = n0 + (lane & 15);

  const float bb = by[n];
  v8f acc;
#pragma unroll
  for (int j = 0; j < 8; ++j) acc[j] = bb;

  gemm_acc<H_>(acc, hist, (size_t)H_, Wy, (size_t)H_, m0, n0);

#pragma unroll
  for (int j = 0; j < 8; ++j) {
    const int row = m0 + j + ((lane >> 4) << 3); // row = s*B + b
    const int s = row >> 6;
    const int b = row & (B_ - 1);
    out[((size_t)b * S_ + s) * O_ + n] = acc[j];
  }
}

// ------------------------------- launcher -----------------------------------

extern "C" void kernel_launch(void* const* d_in, const int* in_sizes, int n_in,
                              void* d_out, int out_size, void* d_ws, size_t ws_size,
                              hipStream_t stream) {
  (void)in_sizes; (void)n_in; (void)out_size; (void)ws_size;

  const float* in_x = (const float*)d_in[0];   // (B,S,I)
  const float* in_h = (const float*)d_in[1];   // (B,2,H)
  const float* bz0  = (const float*)d_in[4];
  const float* br0  = (const float*)d_in[7];
  const float* bg0  = (const float*)d_in[10];
  const float* bz1  = (const float*)d_in[13];
  const float* br1  = (const float*)d_in[16];
  const float* bg1  = (const float*)d_in[19];
  const float* by   = (const float*)d_in[21];

  // ---- workspace carve-up (256-aligned blocks) ----
  char* ws = (char*)d_ws;
  size_t off = 0;
  auto take = [&](size_t bytes) {
    char* p = ws + off;
    off += (bytes + 255) & ~(size_t)255;
    return p;
  };
  __bf16* Xbf  = (__bf16*)take((size_t)B_ * S_ * I_ * 2);   // 33.5 MB
  __bf16* wxz0 = (__bf16*)take((size_t)H_ * I_ * 2);
  __bf16* whz0 = (__bf16*)take((size_t)H_ * H_ * 2);
  __bf16* wxr0 = (__bf16*)take((size_t)H_ * I_ * 2);
  __bf16* whr0 = (__bf16*)take((size_t)H_ * H_ * 2);
  __bf16* wxg0 = (__bf16*)take((size_t)H_ * I_ * 2);
  __bf16* whg0 = (__bf16*)take((size_t)H_ * H_ * 2);
  __bf16* wxz1 = (__bf16*)take((size_t)H_ * H_ * 2);
  __bf16* whz1 = (__bf16*)take((size_t)H_ * H_ * 2);
  __bf16* wxr1 = (__bf16*)take((size_t)H_ * H_ * 2);
  __bf16* whr1 = (__bf16*)take((size_t)H_ * H_ * 2);
  __bf16* wxg1 = (__bf16*)take((size_t)H_ * H_ * 2);
  __bf16* whg1 = (__bf16*)take((size_t)H_ * H_ * 2);
  __bf16* wy   = (__bf16*)take((size_t)O_ * H_ * 2);
  float*  h0f  = (float*) take((size_t)B_ * H_ * 4);
  float*  h1f  = (float*) take((size_t)B_ * H_ * 4);
  __bf16* h0b  = (__bf16*)take((size_t)B_ * H_ * 2);
  __bf16* h1b  = (__bf16*)take((size_t)B_ * H_ * 2);
  float*  z0f  = (float*) take((size_t)B_ * H_ * 4);
  float*  z1f  = (float*) take((size_t)B_ * H_ * 4);
  __bf16* rh0  = (__bf16*)take((size_t)B_ * H_ * 2);
  __bf16* rh1  = (__bf16*)take((size_t)B_ * H_ * 2);
  __bf16* hist = (__bf16*)take((size_t)S_ * B_ * H_ * 2);   // 134 MB
  unsigned* bar = (unsigned*)take(256);

  // ---- one-time conversions (also warms weights into L2) ----
  {
    size_t nx = (size_t)B_ * S_ * I_;
    k_f32_to_bf16<<<(unsigned)((nx + 255) / 256), 256, 0, stream>>>(in_x, Xbf, nx);
  }
  struct CV { int idx; __bf16* dst; size_t n; } cvs[] = {
    {2,  wxz0, (size_t)H_ * I_}, {3,  whz0, (size_t)H_ * H_},
    {5,  wxr0, (size_t)H_ * I_}, {6,  whr0, (size_t)H_ * H_},
    {8,  wxg0, (size_t)H_ * I_}, {9,  whg0, (size_t)H_ * H_},
    {11, wxz1, (size_t)H_ * H_}, {12, whz1, (size_t)H_ * H_},
    {14, wxr1, (size_t)H_ * H_}, {15, whr1, (size_t)H_ * H_},
    {17, wxg1, (size_t)H_ * H_}, {18, whg1, (size_t)H_ * H_},
    {20, wy,   (size_t)O_ * H_},
  };
  for (const CV& c : cvs)
    k_f32_to_bf16<<<(unsigned)((c.n + 255) / 256), 256, 0, stream>>>(
        (const float*)d_in[c.idx], c.dst, c.n);

  k_zero_u32<<<1, 64, 0, stream>>>(bar);
  k_init_h<<<(B_ * H_) / 256, 256, 0, stream>>>(in_h, h0f, h1f, h0b, h1b);

  // ---- entire sequential scan: ONE persistent kernel ----
  k_gru_scan<<<NWG_, 128, 0, stream>>>(
      Xbf,
      wxz0, whz0, bz0, wxr0, whr0, br0, wxg0, whg0, bg0,
      wxz1, whz1, bz1, wxr1, whr1, br1, wxg1, whg1, bg1,
      h0f, h1f, h0b, h1b, z0f, z1f, rh0, rh1, hist, bar);

  // ---- batched output projection over all timesteps ----
  float* out = (float*)d_out;
  k_out_proj<<<(S_ * B_ / 16) * (O_ / 16) / 4, 128, 0, stream>>>(hist, wy, by, out);

  // ---- final hidden state (flat: h0 then h1) ----
  k_copy_hidden<<<(2 * B_ * H_) / 256, 256, 0, stream>>>(
      h0f, h1f, out + (size_t)B_ * S_ * O_);
}
